// LegislativeGraphEncoder_13065290515084
// MI455X (gfx1250) — matrix operand order repaired and checked by reading, not compiled, not run on hardware
//
#include <hip/hip_runtime.h>
#include <math.h>

typedef __attribute__((ext_vector_type(16))) __bf16 bf16x16;
typedef __attribute__((ext_vector_type(8)))  __bf16 bf16x8;
typedef __attribute__((ext_vector_type(8)))  float  f32x8;
typedef __attribute__((ext_vector_type(4)))  float  f32x4;

#define HID 192
#define LT_OFF 205200   // legislator_term node offset
#define BV_OFF 50000    // bill_version node offset

// ---------------------------------------------------------------- helpers

__device__ __forceinline__ float gelu_f(float x) {
  return 0.5f * x * (1.0f + erff(x * 0.70710678118654752f));
}

__device__ __forceinline__ void atomicMaxF(float* addr, float val) {
  unsigned* ua = (unsigned*)addr;
  unsigned old = *ua;
  while (__uint_as_float(old) < val) {
    unsigned assumed = old;
    old = atomicCAS(ua, assumed, __float_as_uint(val));
    if (old == assumed) break;
  }
}

__device__ __forceinline__ f32x8 wmma_bf16(bf16x16 a, bf16x16 b, f32x8 c) {
  return __builtin_amdgcn_wmma_f32_16x16x32_bf16(false, a, false, b, (short)0, c,
                                                 false, false);
}

// A fragment (16x32 bf16, MxK) from LDS row-major [16][ldk], ldk % 32 == 0.
// ISA 7.12.2: lane L holds row L&15; K chunks [8h,8h+8) and [16+8h,16+8h+8).
// Both chunks are 16B-aligned contiguous -> two ds_load_b128.
__device__ __forceinline__ bf16x16 frag_a_lds(const __bf16* sA, int ldk, int k0, int lane) {
  const __bf16* p = sA + (lane & 15) * ldk + k0 + 8 * (lane >> 4);
  bf16x8 lo = *(const bf16x8*)p;
  bf16x8 hi = *(const bf16x8*)(p + 16);
  bf16x16 a;
#pragma unroll
  for (int j = 0; j < 8; ++j) { a[j] = lo[j]; a[8 + j] = hi[j]; }
  return a;
}

// A fragment from global fp32 row-major [M][lda] (lda % 32 == 0, rows padded),
// converted to bf16 on the fly. Four global_load_b128 per fragment.
__device__ __forceinline__ bf16x16 frag_a_glbf(const float* A, long lda, int m0, int k0,
                                               int lane) {
  const float* p = A + (long)(m0 + (lane & 15)) * lda + k0 + 8 * (lane >> 4);
  f32x4 c0 = *(const f32x4*)p;
  f32x4 c1 = *(const f32x4*)(p + 4);
  f32x4 c2 = *(const f32x4*)(p + 16);
  f32x4 c3 = *(const f32x4*)(p + 20);
  bf16x16 a;
#pragma unroll
  for (int j = 0; j < 4; ++j) {
    a[j]      = (__bf16)c0[j];
    a[4 + j]  = (__bf16)c1[j];
    a[8 + j]  = (__bf16)c2[j];
    a[12 + j] = (__bf16)c3[j];
  }
  return a;
}

// B fragment (32x16 bf16, KxN) from global bf16 weight stored [N][ldk], ldk % 32 == 0
// (zero-padded at conversion). Lane L holds col L&15, K chunks [8h,8h+8) and [16+8h,16+8h+8):
// two 16-byte chunks -> two global_load_b128.
__device__ __forceinline__ bf16x16 frag_b_glb(const __bf16* W, long ldk, int n0, int k0,
                                              int lane) {
  const __bf16* p = W + (long)(n0 + (lane & 15)) * ldk + k0 + 8 * (lane >> 4);
  bf16x8 lo = *(const bf16x8*)p;
  bf16x8 hi = *(const bf16x8*)(p + 16);
  bf16x16 b;
#pragma unroll
  for (int j = 0; j < 8; ++j) { b[j] = lo[j]; b[8 + j] = hi[j]; }
  return b;
}

// ---------------------------------------------------------------- utility kernels

// fp32 [N][K] -> bf16 [N][Kp], zero-padded K..Kp (Kp % 32 == 0)
__global__ void k_cvt_pad(const float* __restrict__ s, __bf16* __restrict__ d,
                          int N, int K, int Kp) {
  int i = blockIdx.x * 256 + threadIdx.x;
  if (i >= N * Kp) return;
  int n = i / Kp, k = i - n * Kp;
  d[i] = (k < K) ? (__bf16)s[(long)n * K + k] : (__bf16)0.f;
}

__global__ void k_fill(float* p, float v, long n) {
  long i = (long)blockIdx.x * 256 + threadIdx.x;
  if (i < n) p[i] = v;
}

__global__ void k_add(float* d, const float* s, long n) {
  long i = (long)blockIdx.x * 256 + threadIdx.x;
  if (i < n) d[i] += s[i];
}

// ---------------------------------------------------------------- input projection
__device__ __forceinline__ float proj_val(const float* xr, float t, const float* w0,
                                          const float* w, const float* tb, int din, int k) {
  if (k < din) return xr[k];
  if (k == din) return w0[0] * t + tb[0];
  return __sinf(w[k - din - 1] * t + tb[k - din]);
}

// h[nt] = gelu( LN(x ++ t2v(t), g, b) @ W.T ),  W bf16 [192][kpad]
__global__ void __launch_bounds__(128)
k_proj(const float* __restrict__ x, const float* __restrict__ tsv,
       const float* __restrict__ w0, const float* __restrict__ w,
       const float* __restrict__ tb, const float* __restrict__ ln_g,
       const float* __restrict__ ln_b, const __bf16* __restrict__ Wb,
       float* __restrict__ hout, int nn, int din, int dtot, int kpad) {
  __shared__ __bf16 sA[16 * 800];
  const int m0 = blockIdx.x * 16;
  const int tid = threadIdx.x;
  // stage 0: 8 lanes per row build LN(x ++ t2v) in bf16
  {
    const int row = tid >> 3, part = tid & 7;
    const int node = m0 + row;
    __bf16* srow = sA + row * kpad;
    const int chunk = kpad >> 3;
    const int kb = part * chunk, ke = kb + chunk;
    if (node < nn) {
      const float* xr = x + (long)node * din;
      const float t = tsv ? tsv[node] : 0.f;
      float s1 = 0.f, s2 = 0.f;
      for (int k = kb; k < ke && k < dtot; ++k) {
        const float v = proj_val(xr, t, w0, w, tb, din, k);
        s1 += v; s2 += v * v;
      }
#pragma unroll
      for (int mm = 1; mm < 8; mm <<= 1) {
        s1 += __shfl_xor(s1, mm, 32);
        s2 += __shfl_xor(s2, mm, 32);
      }
      const float mu  = s1 / dtot;
      const float var = fmaxf(s2 / dtot - mu * mu, 0.f);
      const float inv = rsqrtf(var + 1e-5f);
      for (int k = kb; k < ke; ++k) {
        if (k < dtot) {
          const float v = proj_val(xr, t, w0, w, tb, din, k);
          srow[k] = (__bf16)(((v - mu) * inv) * ln_g[k] + ln_b[k]);
        } else {
          srow[k] = (__bf16)0.f;
        }
      }
    } else {
      for (int k = kb; k < ke; ++k) srow[k] = (__bf16)0.f;
    }
  }
  __syncthreads();
  const int wave = tid >> 5, lane = tid & 31;
  for (int tt = wave; tt < 12; tt += 4) {       // 12 x 16 = 192 output cols
    const int n0 = tt * 16;
    f32x8 acc = {};
    for (int k0 = 0; k0 < kpad; k0 += 32)
      acc = wmma_bf16(frag_a_lds(sA, kpad, k0, lane),
                      frag_b_glb(Wb, kpad, n0, k0, lane), acc);
    const int n = n0 + (lane & 15), hh = lane >> 4;
#pragma unroll
    for (int i = 0; i < 8; ++i) {
      const int m = m0 + 8 * hh + i;
      if (m < nn) hout[(long)m * HID + n] = gelu_f(acc[i]);
    }
  }
}

// ---------------------------------------------------------------- generic GEMM (QKV)
// C[M,N] = A[M,K] @ W[N,K]^T ; A fp32 rows padded to 16-row multiple, K % 32 == 0.
__global__ void __launch_bounds__(128)
k_gemm(const float* __restrict__ A, const __bf16* __restrict__ W,
       float* __restrict__ C, int M, int N, int K) {
  const int m0 = blockIdx.x * 16;
  const int wave = threadIdx.x >> 5, lane = threadIdx.x & 31;
  const int n0 = blockIdx.y * 64 + wave * 16;
  if (n0 >= N) return;
  // gfx1250 prefetch path: pull the next M-tile's A rows toward L2
  __builtin_prefetch((const void*)(A + (long)(m0 + 16) * K), 0, 1);
  f32x8 acc = {};
  for (int k0 = 0; k0 < K; k0 += 32)
    acc = wmma_bf16(frag_a_glbf(A, K, m0, k0, lane),
                    frag_b_glb(W, K, n0, k0, lane), acc);
  const int n = n0 + (lane & 15), hh = lane >> 4;
#pragma unroll
  for (int i = 0; i < 8; ++i) {
    const int m = m0 + 8 * hh + i;
    if (m < M) C[(long)m * N + n] = acc[i];
  }
}

// ---------------------------------------------------------------- edge attention
__global__ void k_attn_logits(const float* __restrict__ Q, const float* __restrict__ Kb,
                              const float* __restrict__ R, const int* __restrict__ es,
                              const int* __restrict__ et, float* __restrict__ logit,
                              float* __restrict__ maxb, int ne, int soff, int doff) {
  long gid = (long)blockIdx.x * 256 + threadIdx.x;
  if (gid >= (long)ne * 4) return;
  const int e = (int)(gid >> 2), hd = (int)(gid & 3);
  const int s = soff + es[e], t = doff + et[e];
  const f32x4* q = (const f32x4*)(Q + (long)t * HID + hd * 48);
  const f32x4* k = (const f32x4*)(Kb + (long)s * HID + hd * 48);
  const f32x4* r = (const f32x4*)(R + hd * 48);
  float l = 0.f;
#pragma unroll
  for (int i = 0; i < 12; ++i) {
    const f32x4 qq = q[i], kk = k[i], rr = r[i];
    l += qq[0] * (kk[0] + rr[0]) + qq[1] * (kk[1] + rr[1]) +
         qq[2] * (kk[2] + rr[2]) + qq[3] * (kk[3] + rr[3]);
  }
  l *= 0.14433756729740643f;               // 1/sqrt(48)
  l = (l > 0.f) ? l : 0.01f * l;           // leaky_relu
  logit[gid] = l;
  atomicMaxF(&maxb[(long)t * 4 + hd], l);
}

__global__ void k_attn_exp(const float* __restrict__ logit, const float* __restrict__ maxb,
                           const int* __restrict__ et, float* __restrict__ ebuf,
                           float* __restrict__ sumb, int ne, int doff) {
  long gid = (long)blockIdx.x * 256 + threadIdx.x;
  if (gid >= (long)ne * 4) return;
  const int e = (int)(gid >> 2), hd = (int)(gid & 3);
  const int t = doff + et[e];
  const float ev = __expf(logit[gid] - maxb[(long)t * 4 + hd]);
  ebuf[gid] = ev;
  atomicAdd(&sumb[(long)t * 4 + hd], ev);
}

__global__ void k_attn_scatter(const float* __restrict__ ebuf, const float* __restrict__ sumb,
                               const float* __restrict__ V, const int* __restrict__ es,
                               const int* __restrict__ et, const float* __restrict__ ew,
                               float* __restrict__ msg, int ne, int soff, int doff) {
  long gid = (long)blockIdx.x * 256 + threadIdx.x;
  if (gid >= (long)ne * 4) return;
  const int e = (int)(gid >> 2), hd = (int)(gid & 3);
  const int s = soff + es[e], t = doff + et[e];
  float a = ebuf[gid] / (sumb[(long)t * 4 + hd] + 1e-16f);
  if (ew) a *= (1.f + ew[e]);
  const f32x4* v4 = (const f32x4*)(V + (long)s * HID + hd * 48);
  float* mrow = msg + (long)t * HID + hd * 48;
#pragma unroll
  for (int i = 0; i < 12; ++i) {
    const f32x4 vv = v4[i];
    atomicAdd(&mrow[4 * i + 0], a * vv[0]);
    atomicAdd(&mrow[4 * i + 1], a * vv[1]);
    atomicAdd(&mrow[4 * i + 2], a * vv[2]);
    atomicAdd(&mrow[4 * i + 3], a * vv[3]);
  }
}

// ---------------------------------------------------------------- fused residual+LN+FFN
// h = hr + gelu(LN2(hr) @ W1^T) @ W2^T,  hr = h + msg ; 768-wide stays in LDS.
__global__ void __launch_bounds__(128)
k_ffn(float* __restrict__ h, const float* __restrict__ msg,
      const __bf16* __restrict__ W1, const __bf16* __restrict__ W2,
      const float* __restrict__ g, const float* __restrict__ b, int M) {
  __shared__ float  sHR[16 * HID];
  __shared__ __bf16 sA[16 * HID];
  __shared__ __bf16 sU[16 * 768];
  const int m0 = blockIdx.x * 16;
  const int tid = threadIdx.x;
  // stage 0: 8 lanes per row; residual + double-LN -> bf16
  {
    const int row = tid >> 3, part = tid & 7;
    const int m = m0 + row;
    float*  hr   = sHR + row * HID;
    __bf16* arow = sA + row * HID;
    const int jb = part * 6, je = jb + 6;      // 6 f32x4 chunks = 24 floats
    if (m < M) {
      const f32x4* hv = (const f32x4*)(h + (long)m * HID);
      const f32x4* mg = (const f32x4*)(msg + (long)m * HID);
      f32x4* hrv = (f32x4*)hr;
      float s1 = 0.f, s2 = 0.f;
      for (int j = jb; j < je; ++j) {
        const f32x4 v = hv[j] + mg[j];
        hrv[j] = v;
        s1 += v[0] + v[1] + v[2] + v[3];
        s2 += v[0] * v[0] + v[1] * v[1] + v[2] * v[2] + v[3] * v[3];
      }
#pragma unroll
      for (int mm = 1; mm < 8; mm <<= 1) {
        s1 += __shfl_xor(s1, mm, 32);
        s2 += __shfl_xor(s2, mm, 32);
      }
      const float mu  = s1 / HID;
      const float var = fmaxf(s2 / HID - mu * mu, 0.f);
      const float inv = rsqrtf(var + 1e-5f);
      float u1 = 0.f, u2 = 0.f;                 // second LN stats
      for (int j = jb; j < je; ++j) {
        const f32x4 v = hrv[j];
#pragma unroll
        for (int c = 0; c < 4; ++c) {
          const float y = (v[c] - mu) * inv;
          u1 += y; u2 += y * y;
        }
      }
#pragma unroll
      for (int mm = 1; mm < 8; mm <<= 1) {
        u1 += __shfl_xor(u1, mm, 32);
        u2 += __shfl_xor(u2, mm, 32);
      }
      const float mu2  = u1 / HID;
      const float var2 = fmaxf(u2 / HID - mu2 * mu2, 0.f);
      const float inv2 = rsqrtf(var2 + 1e-5f);
      for (int j = jb; j < je; ++j) {
        const f32x4 v = hrv[j];
#pragma unroll
        for (int c = 0; c < 4; ++c) {
          const float y = (v[c] - mu) * inv;
          arow[4 * j + c] = (__bf16)(((y - mu2) * inv2) * g[4 * j + c] + b[4 * j + c]);
        }
      }
    } else {
      for (int j = jb; j < je; ++j) {
        ((f32x4*)hr)[j] = (f32x4){0.f, 0.f, 0.f, 0.f};
#pragma unroll
        for (int c = 0; c < 4; ++c) arow[4 * j + c] = (__bf16)0.f;
      }
    }
  }
  __syncthreads();
  const int wave = tid >> 5, lane = tid & 31;
  for (int tt = wave; tt < 48; tt += 4) {       // 768 = 48 tiles of 16
    const int n0 = tt * 16;
    f32x8 acc = {};
#pragma unroll
    for (int k0 = 0; k0 < HID; k0 += 32)
      acc = wmma_bf16(frag_a_lds(sA, HID, k0, lane),
                      frag_b_glb(W1, HID, n0, k0, lane), acc);
    const int n = n0 + (lane & 15), hh = lane >> 4;
#pragma unroll
    for (int i = 0; i < 8; ++i)
      sU[(8 * hh + i) * 768 + n] = (__bf16)gelu_f(acc[i]);
  }
  __syncthreads();
  for (int tt = wave; tt < 12; tt += 4) {
    const int n0 = tt * 16;
    f32x8 acc = {};
#pragma unroll
    for (int k0 = 0; k0 < 768; k0 += 32)
      acc = wmma_bf16(frag_a_lds(sU, 768, k0, lane),
                      frag_b_glb(W2, 768, n0, k0, lane), acc);
    const int n = n0 + (lane & 15), hh = lane >> 4;
#pragma unroll
    for (int i = 0; i < 8; ++i) {
      const int m = m0 + 8 * hh + i;
      if (m < M) h[(long)m * HID + n] = sHR[(8 * hh + i) * HID + n] + acc[i];
    }
  }
}

// ---------------------------------------------------------------- fused vote-edge MLP
// ef = (relu(raw@W1^T+b1)@W2^T+b2)*(clip(pol)+.01); msg[bv t] += h[lt s]*ef
__global__ void __launch_bounds__(128)
k_vote(const float* __restrict__ va, const int* __restrict__ es, const int* __restrict__ et,
       const __bf16* __restrict__ W1, const __bf16* __restrict__ W2,
       const float* __restrict__ b1, const float* __restrict__ b2,
       const float* __restrict__ h, float* __restrict__ msg, int ne) {
  __shared__ __bf16 sR[16 * 384];
  __shared__ __bf16 sT[16 * HID];
  __shared__ float  sScale[16];
  __shared__ int    sS[16], sD[16], sV[16];
  const int e0 = blockIdx.x * 16;
  const int tid = threadIdx.x;
  {
    const int row = tid >> 3, part = tid & 7;
    const int e = e0 + row;
    __bf16* rr = sR + row * 384;
    const int kb = part * 48, ke = kb + 48;
    if (e < ne) {
      const float* r = va + (long)e * 385;
      if (part == 0) {
        const float pol = fminf(fmaxf(r[0], 0.f), 1.f);
        sScale[row] = pol + 0.01f;
        sS[row] = es[e]; sD[row] = et[e]; sV[row] = 1;
      }
      for (int k = kb; k < ke; ++k) rr[k] = (__bf16)r[1 + k];
    } else {
      if (part == 0) { sScale[row] = 0.f; sS[row] = 0; sD[row] = 0; sV[row] = 0; }
      for (int k = kb; k < ke; ++k) rr[k] = (__bf16)0.f;
    }
  }
  __syncthreads();
  const int wave = tid >> 5, lane = tid & 31;
  for (int tt = wave; tt < 12; tt += 4) {
    const int n0 = tt * 16;
    f32x8 acc = {};
#pragma unroll
    for (int k0 = 0; k0 < 384; k0 += 32)
      acc = wmma_bf16(frag_a_lds(sR, 384, k0, lane),
                      frag_b_glb(W1, 384, n0, k0, lane), acc);
    const int n = n0 + (lane & 15), hh = lane >> 4;
#pragma unroll
    for (int i = 0; i < 8; ++i)
      sT[(8 * hh + i) * HID + n] = (__bf16)fmaxf(acc[i] + b1[n], 0.f);
  }
  __syncthreads();
  for (int tt = wave; tt < 12; tt += 4) {
    const int n0 = tt * 16;
    f32x8 acc = {};
#pragma unroll
    for (int k0 = 0; k0 < HID; k0 += 32)
      acc = wmma_bf16(frag_a_lds(sT, HID, k0, lane),
                      frag_b_glb(W2, HID, n0, k0, lane), acc);
    const int n = n0 + (lane & 15), hh = lane >> 4;
#pragma unroll
    for (int i = 0; i < 8; ++i) {
      const int rl = 8 * hh + i;
      if (sV[rl]) {
        const float ef = (acc[i] + b2[n]) * sScale[rl];
        const float hl = h[(long)(LT_OFF + sS[rl]) * HID + n];
        atomicAdd(&msg[(long)(BV_OFF + sD[rl]) * HID + n], hl * ef);
      }
    }
  }
}

// ---------------------------------------------------------------- final LN + relu
__global__ void k_final(const float* __restrict__ h, const float* __restrict__ g,
                        const float* __restrict__ b, float* __restrict__ out, int nn) {
  const int m = blockIdx.x * 256 + threadIdx.x;
  if (m >= nn) return;
  const f32x4* r4 = (const f32x4*)(h + (long)m * HID);
  float s1 = 0.f, s2 = 0.f;
#pragma unroll 4
  for (int j = 0; j < 48; ++j) {
    const f32x4 v = r4[j];
    s1 += v[0] + v[1] + v[2] + v[3];
    s2 += v[0] * v[0] + v[1] * v[1] + v[2] * v[2] + v[3] * v[3];
  }
  const float mu  = s1 / HID;
  const float var = fmaxf(s2 / HID - mu * mu, 0.f);
  const float inv = rsqrtf(var + 1e-5f);
  const f32x4* g4 = (const f32x4*)g;
  const f32x4* b4 = (const f32x4*)b;
  f32x4* o4 = (f32x4*)(out + (long)m * HID);
#pragma unroll 4
  for (int j = 0; j < 48; ++j) {
    const f32x4 v = r4[j], gg = g4[j], bb = b4[j];
    f32x4 o;
#pragma unroll
    for (int c = 0; c < 4; ++c)
      o[c] = fmaxf(((v[c] - mu) * inv) * gg[c] + bb[c], 0.f);
    o4[j] = o;
  }
}

// ---------------------------------------------------------------- host orchestration

extern "C" void kernel_launch(void* const* d_in, const int* in_sizes, int n_in,
                              void* d_out, int out_size, void* d_ws, size_t ws_size,
                              hipStream_t stream) {
  (void)in_sizes; (void)n_in; (void)out_size; (void)ws_size;
  auto F = [&](int i) { return (const float*)d_in[i]; };
  auto I = [&](int i) { return (const int*)d_in[i]; };

  // sorted node-type order: bill, bill_version, committee, donor, legislator_term, lobbyist, session
  static const int NN[7]   = {50000, 150000, 200, 5000, 10000, 2000, 20};
  static const int NOFF[7] = {0, 50000, 200000, 200200, 205200, 215200, 217200};
  static const int XD[7]   = {385, 389, 2, 384, 769, 385, 384};
  static const int TSI[7]  = {80, 81, -1, -1, 82, -1, -1};
  const int NTOT  = 217220;
  const int NROWS = 217232;   // padded to 16-row multiple for unguarded A-frag loads

  char* ws = (char*)d_ws;
  size_t off = 0;
  auto alloc = [&](size_t bytes) {
    void* p = ws + off;
    off += (bytes + 255) & ~(size_t)255;
    return p;
  };

  __bf16* projW[7];
  int dtot[7], kpad[7];
  for (int t = 0; t < 7; ++t) {
    dtot[t] = XD[t] + ((TSI[t] >= 0) ? 12 : 0);
    kpad[t] = (dtot[t] + 31) / 32 * 32;
    projW[t] = (__bf16*)alloc((size_t)192 * kpad[t] * 2);
  }
  __bf16 *Qw[3], *Kw[3], *Vw[3], *W1w[3], *W2w[3];
  for (int l = 0; l < 3; ++l) {
    Kw[l]  = (__bf16*)alloc(36864 * 2);
    Qw[l]  = (__bf16*)alloc(36864 * 2);
    Vw[l]  = (__bf16*)alloc(36864 * 2);
    W1w[l] = (__bf16*)alloc(147456 * 2);
    W2w[l] = (__bf16*)alloc(147456 * 2);
  }
  __bf16* vW1 = (__bf16*)alloc(73728 * 2);
  __bf16* vW2 = (__bf16*)alloc(36864 * 2);

  float* hbuf = (float*)alloc((size_t)NROWS * HID * 4);
  float* msg  = (float*)alloc((size_t)NROWS * HID * 4);
  float* Qb   = (float*)alloc((size_t)NROWS * HID * 4);
  float* Kb   = (float*)alloc((size_t)NROWS * HID * 4);
  float* Vb   = (float*)alloc((size_t)NROWS * HID * 4);
  float* logits = (float*)alloc((size_t)400000 * 4 * 4);
  float* ebuf   = (float*)alloc((size_t)400000 * 4 * 4);
  float* maxb   = (float*)alloc((size_t)NTOT * 4 * 4);
  float* sumb   = (float*)alloc((size_t)NTOT * 4 * 4);

  auto cvt = [&](const float* s, __bf16* d, int N, int K, int Kp) {
    k_cvt_pad<<<(N * Kp + 255) / 256, 256, 0, stream>>>(s, d, N, K, Kp);
  };
  // weight bf16 conversion (pytree order: layer dict = K,Q,V,W1,W2,ffn_ln_b,ffn_ln_g,rel[4])
  for (int t = 0; t < 7; ++t) cvt(F(52 + 3 * t), projW[t], 192, dtot[t], kpad[t]);
  for (int l = 0; l < 3; ++l) {
    const int base = 5 + 11 * l;
    cvt(F(base + 0), Kw[l], 192, 192, 192);
    cvt(F(base + 1), Qw[l], 192, 192, 192);
    cvt(F(base + 2), Vw[l], 192, 192, 192);
    cvt(F(base + 3), W1w[l], 768, 192, 192);
    cvt(F(base + 4), W2w[l], 192, 768, 768);
  }
  cvt(F(76), vW1, 192, 384, 384);
  cvt(F(77), vW2, 192, 192, 192);

  // input projections (t2v: b=73, w=74, w0=75)
  for (int t = 0; t < 7; ++t) {
    const float* tsv = (TSI[t] >= 0) ? F(TSI[t]) : nullptr;
    k_proj<<<(NN[t] + 15) / 16, 128, 0, stream>>>(
        F(84 + t), tsv, F(75), F(74), F(73), F(54 + 3 * t), F(53 + 3 * t),
        projW[t], hbuf + (size_t)NOFF[t] * HID, NN[t], XD[t], dtot[t], kpad[t]);
  }

  struct Rel { int st, dt, ne, eidx, ridx, hw; };
  const Rel rels[4] = {
      {4, 1, 400000, 3, 3, 1},  // legislator_term -> voted_on -> bill_version (+ew)
      {0, 1, 200000, 0, 0, 0},  // bill -> has_version -> bill_version
      {1, 0, 200000, 1, 1, 0},  // bill_version -> version_of -> bill
      {2, 0, 100000, 2, 2, 0},  // committee -> reviews -> bill
  };

  const long nh = (long)NTOT * HID;
  for (int l = 0; l < 3; ++l) {
    const int base = 5 + 11 * l;
    k_fill<<<(int)((nh + 255) / 256), 256, 0, stream>>>(msg, 0.f, nh);
    dim3 gg((NTOT + 15) / 16, 3);
    k_gemm<<<gg, 128, 0, stream>>>(hbuf, Qw[l], Qb, NTOT, HID, HID);
    k_gemm<<<gg, 128, 0, stream>>>(hbuf, Kw[l], Kb, NTOT, HID, HID);
    k_gemm<<<gg, 128, 0, stream>>>(hbuf, Vw[l], Vb, NTOT, HID, HID);
    for (int r = 0; r < 4; ++r) {
      const Rel& R = rels[r];
      const int soff = NOFF[R.st], doff = NOFF[R.dt];
      const long nd4 = (long)NN[R.dt] * 4;
      k_fill<<<(int)((nd4 + 255) / 256), 256, 0, stream>>>(maxb + (size_t)doff * 4, -1e30f, nd4);
      k_fill<<<(int)((nd4 + 255) / 256), 256, 0, stream>>>(sumb + (size_t)doff * 4, 0.f, nd4);
      const int* es = I(R.eidx);
      const int* et = es + R.ne;
      const int gb = (int)(((long)R.ne * 4 + 255) / 256);
      k_attn_logits<<<gb, 256, 0, stream>>>(Qb, Kb, F(base + 7 + R.ridx), es, et,
                                            logits, maxb, R.ne, soff, doff);
      k_attn_exp<<<gb, 256, 0, stream>>>(logits, maxb, et, ebuf, sumb, R.ne, doff);
      k_attn_scatter<<<gb, 256, 0, stream>>>(ebuf, sumb, Vb, es, et,
                                             R.hw ? F(4) : nullptr, msg, R.ne, soff, doff);
    }
    k_ffn<<<(NTOT + 15) / 16, 128, 0, stream>>>(hbuf, msg, W1w[l], W2w[l],
                                                F(base + 6), F(base + 5), NTOT);
  }

  // vote-edge MLP scatter into bill_version
  k_fill<<<(int)((nh + 255) / 256), 256, 0, stream>>>(msg, 0.f, nh);
  {
    const int* es = I(3);
    const int* et = es + 400000;
    k_vote<<<(400000 + 15) / 16, 128, 0, stream>>>(F(83), es, et, vW1, vW2,
                                                   F(78), F(79), hbuf, msg, 400000);
  }
  {
    const long nb = (long)150000 * HID;
    k_add<<<(int)((nb + 255) / 256), 256, 0, stream>>>(hbuf + (size_t)BV_OFF * HID,
                                                       msg + (size_t)BV_OFF * HID, nb);
  }

  // final LN + relu, directly into d_out (sorted node-type concatenation)
  for (int t = 0; t < 7; ++t) {
    k_final<<<(NN[t] + 255) / 256, 256, 0, stream>>>(
        hbuf + (size_t)NOFF[t] * HID, F(39 + 2 * t), F(38 + 2 * t),
        (float*)d_out + (size_t)NOFF[t] * HID, NN[t]);
  }
}
